// LSTMModel_89584427860119
// MI455X (gfx1250) — matrix operand order repaired and checked by reading, not compiled, not run on hardware
//
#include <hip/hip_runtime.h>
#include <hip/hip_bf16.h>

#define B_SZ 2048
#define T_SZ 512
#define D_SZ 64
#define H_SZ 128
#define G_SZ 512   // 4*H
#define FC_SZ 64
#define O_SZ 5

typedef __bf16 bf16x16 __attribute__((ext_vector_type(16)));
typedef float f32x8 __attribute__((ext_vector_type(8)));

union BfPack {
    bf16x16 v;
    unsigned u[8];
};

#if __has_builtin(__builtin_amdgcn_tanhf)
#define TANHF(x) __builtin_amdgcn_tanhf(x)
#elif __has_builtin(__builtin_amdgcn_tanh_f32)
#define TANHF(x) __builtin_amdgcn_tanh_f32(x)
#else
#define TANHF(x) tanhf(x)
#endif

__device__ __forceinline__ float sigmoid_via_tanh(float x) {
    // sigmoid(x) = 0.5 * tanh(0.5x) + 0.5  -> single TRANS op + FMA
    return __builtin_fmaf(0.5f, TANHF(0.5f * x), 0.5f);
}

// ---------------------------------------------------------------------------
// Bulk f32 -> bf16 conversion (bandwidth bound, hardware v_cvt)
// ---------------------------------------------------------------------------
__global__ __launch_bounds__(256)
void f32_to_bf16_kernel(const float* __restrict__ in, __bf16* __restrict__ out, unsigned n) {
    unsigned i = blockIdx.x * blockDim.x + threadIdx.x;
    const unsigned stride = gridDim.x * blockDim.x;
    for (; i < n; i += stride) out[i] = (__bf16)in[i];
}

// ---------------------------------------------------------------------------
// One block = 16 batch rows for ALL timesteps. 8 waves; wave w owns the
// 16-wide H-slice [w*16, w*16+16). h_{t-1} is shared via a 4KB LDS stage.
// Weights cached in LDS as bf16 row-major [col][k]: B-operand K-pairs are
// aligned ds_read_b32 (merged to ds_load_b128 by the backend).
// ---------------------------------------------------------------------------
template <int DIN, bool WRITE_SEQ>
__global__ __launch_bounds__(256, 1)
void lstm_layer_kernel(const __bf16* __restrict__ Xb,          // (B,T,DIN) bf16
                       const float* __restrict__ Wih,          // (4H, DIN)
                       const float* __restrict__ Whh,          // (4H, H)
                       const float* __restrict__ bih,          // (4H)
                       const float* __restrict__ bhh,          // (4H)
                       __bf16* __restrict__ HseqOut,           // (B,T,H) bf16 if WRITE_SEQ
                       float* __restrict__ HlastOut)           // (B,H) f32 if !WRITE_SEQ
{
    __shared__ alignas(16) __bf16 whh_lds[G_SZ * H_SZ];   // 128 KB
    __shared__ alignas(16) __bf16 wih_lds[G_SZ * DIN];    // 64/128 KB
    __shared__ alignas(16) __bf16 hstage[16 * H_SZ];      // 4 KB
    __shared__ alignas(16) float  bias_lds[G_SZ];         // 2 KB

    const int tid  = threadIdx.x;
    const int lane = tid & 31;
    const int wv   = tid >> 5;       // 0..7 : which 16-wide H slice
    const int row0 = blockIdx.x * 16;
    const int mlo  = lane & 15;
    const int hi   = lane >> 4;      // lane half

    // ---- one-time: weights -> bf16 LDS (hardware cvt), fold biases ----
    for (int i = tid; i < G_SZ * H_SZ; i += 256) whh_lds[i] = (__bf16)Whh[i];
    for (int i = tid; i < G_SZ * DIN;  i += 256) wih_lds[i] = (__bf16)Wih[i];
    for (int i = tid; i < G_SZ;        i += 256) bias_lds[i] = bih[i] + bhh[i];
    for (int i = tid; i < 16 * H_SZ;   i += 256) hstage[i] = (__bf16)0.0f;   // h_0 = 0
    __syncthreads();

    f32x8 c = {0.f, 0.f, 0.f, 0.f, 0.f, 0.f, 0.f, 0.f};   // c_0 = 0 (this wave's tile)

    const float bi = bias_lds[0 * H_SZ + wv * 16 + mlo];
    const float bf = bias_lds[1 * H_SZ + wv * 16 + mlo];
    const float bg = bias_lds[2 * H_SZ + wv * 16 + mlo];
    const float bo = bias_lds[3 * H_SZ + wv * 16 + mlo];
    const f32x8 cinit[4] = {
        {bi, bi, bi, bi, bi, bi, bi, bi},
        {bf, bf, bf, bf, bf, bf, bf, bf},
        {bg, bg, bg, bg, bg, bg, bg, bg},
        {bo, bo, bo, bo, bo, bo, bo, bo},
    };

    for (int t = 0; t < T_SZ; ++t) {
        // ---- A operand: h_{t-1} from LDS stage (16x128 bf16) ----
        BfPack ah[H_SZ / 32];
#pragma unroll
        for (int kc = 0; kc < H_SZ / 32; ++kc) {
#pragma unroll
            for (int v = 0; v < 8; ++v) {
                const int k = kc * 32 + ((v < 4) ? 2 * v : 16 + 2 * (v - 4)) + (hi ? 8 : 0);
                ah[kc].u[v] = *(const unsigned*)&hstage[mlo * H_SZ + k];
            }
        }
        // ---- A operand: x_t (16xDIN bf16, direct b32 gathers) ----
        BfPack ax[DIN / 32];
#pragma unroll
        for (int kc = 0; kc < DIN / 32; ++kc) {
#pragma unroll
            for (int v = 0; v < 8; ++v) {
                const int k = kc * 32 + ((v < 4) ? 2 * v : 16 + 2 * (v - 4)) + (hi ? 8 : 0);
                ax[kc].u[v] = *(const unsigned*)&Xb[((size_t)(row0 + mlo) * T_SZ + (size_t)t) * DIN + k];
            }
        }
        __syncthreads();   // everyone done reading h_{t-1}

        // ---- gates i,f,g,o for this wave's 16 columns (bias folded into C) ----
        f32x8 gacc[4];
#pragma unroll
        for (int g = 0; g < 4; ++g) {
            const int col = g * H_SZ + wv * 16 + mlo;   // B-operand column for this lane
            f32x8 acc = cinit[g];
            // recurrent part: h_{t-1} @ Whh^T  (K = 128)
#pragma unroll
            for (int kc = 0; kc < H_SZ / 32; ++kc) {
                BfPack b;
#pragma unroll
                for (int v = 0; v < 8; ++v) {
                    const int k = kc * 32 + ((v < 4) ? 2 * v : 16 + 2 * (v - 4)) + (hi ? 8 : 0);
                    b.u[v] = *(const unsigned*)&whh_lds[col * H_SZ + k];
                }
                acc = __builtin_amdgcn_wmma_f32_16x16x32_bf16(
                    false, ah[kc].v, false, b.v, (short)0, acc, false, false);
            }
            // input part: x_t @ Wih^T  (K = DIN)
#pragma unroll
            for (int kc = 0; kc < DIN / 32; ++kc) {
                BfPack b;
#pragma unroll
                for (int v = 0; v < 8; ++v) {
                    const int k = kc * 32 + ((v < 4) ? 2 * v : 16 + 2 * (v - 4)) + (hi ? 8 : 0);
                    b.u[v] = *(const unsigned*)&wih_lds[col * DIN + k];
                }
                acc = __builtin_amdgcn_wmma_f32_16x16x32_bf16(
                    false, ax[kc].v, false, b.v, (short)0, acc, false, false);
            }
            gacc[g] = acc;
        }

        // ---- elementwise LSTM cell update (hardware tanh) ----
        f32x8 hnew;
#pragma unroll
        for (int r = 0; r < 8; ++r) {
            const float iv = sigmoid_via_tanh(gacc[0][r]);
            const float fv = sigmoid_via_tanh(gacc[1][r]);
            const float gv = TANHF(gacc[2][r]);
            const float ov = sigmoid_via_tanh(gacc[3][r]);
            const float cv = __builtin_fmaf(fv, c[r], iv * gv);
            c[r] = cv;
            hnew[r] = ov * TANHF(cv);
        }

        // ---- publish h_t (C/D layout: row = r + 8*hi, col = wv*16 + mlo) ----
#pragma unroll
        for (int r = 0; r < 8; ++r) {
            const int m = r + hi * 8;
            const int n = wv * 16 + mlo;
            const __bf16 hb = (__bf16)hnew[r];
            hstage[m * H_SZ + n] = hb;
            if (WRITE_SEQ) {
                HseqOut[((size_t)(row0 + m) * T_SZ + (size_t)t) * H_SZ + n] = hb;
            }
        }
        __syncthreads();   // h_t visible to all waves before next step
    }

    if (!WRITE_SEQ) {
        for (int i = tid; i < 16 * H_SZ; i += 256) {
            const int m = i / H_SZ, k = i % H_SZ;
            HlastOut[(size_t)(row0 + m) * H_SZ + k] = (float)hstage[i];
        }
    }
}

// ---------------------------------------------------------------------------
// Tiny FC head: out = relu(h @ W1^T + b1) @ W2^T + b2   (~16 MMAC total)
// ---------------------------------------------------------------------------
__global__ __launch_bounds__(256)
void fc_head_kernel(const float* __restrict__ Hlast,
                    const float* __restrict__ W1, const float* __restrict__ b1,
                    const float* __restrict__ W2, const float* __restrict__ b2,
                    float* __restrict__ Out)
{
    const int b = blockIdx.x * blockDim.x + threadIdx.x;
    if (b >= B_SZ) return;
    float o[O_SZ];
#pragma unroll
    for (int j = 0; j < O_SZ; ++j) o[j] = b2[j];
    for (int f = 0; f < FC_SZ; ++f) {
        float z = b1[f];
        for (int k = 0; k < H_SZ; ++k) z += Hlast[(size_t)b * H_SZ + k] * W1[f * H_SZ + k];
        z = fmaxf(z, 0.f);
#pragma unroll
        for (int j = 0; j < O_SZ; ++j) o[j] += z * W2[j * FC_SZ + f];
    }
#pragma unroll
    for (int j = 0; j < O_SZ; ++j) Out[(size_t)b * O_SZ + j] = o[j];
}

extern "C" void kernel_launch(void* const* d_in, const int* in_sizes, int n_in,
                              void* d_out, int out_size, void* d_ws, size_t ws_size,
                              hipStream_t stream) {
    const float* x    = (const float*)d_in[0];
    const float* Wih0 = (const float*)d_in[1];
    const float* Whh0 = (const float*)d_in[2];
    const float* bih0 = (const float*)d_in[3];
    const float* bhh0 = (const float*)d_in[4];
    const float* Wih1 = (const float*)d_in[5];
    const float* Whh1 = (const float*)d_in[6];
    const float* bih1 = (const float*)d_in[7];
    const float* bhh1 = (const float*)d_in[8];
    const float* W1   = (const float*)d_in[9];
    const float* b1   = (const float*)d_in[10];
    const float* W2   = (const float*)d_in[11];
    const float* b2   = (const float*)d_in[12];

    // workspace layout:
    //   h0seq : B*T*H bf16   (256 MB)
    //   xbf   : B*T*D bf16   (128 MB)
    //   hlast : B*H   f32    (1 MB)
    __bf16* h0seq = (__bf16*)d_ws;
    __bf16* xbf   = (__bf16*)((char*)d_ws + (size_t)B_SZ * T_SZ * H_SZ * sizeof(__bf16));
    float*  hlast = (float*)((char*)d_ws + (size_t)B_SZ * T_SZ * (H_SZ + D_SZ) * sizeof(__bf16));

    const unsigned nx = B_SZ * T_SZ * D_SZ;
    f32_to_bf16_kernel<<<2048, 256, 0, stream>>>(x, xbf, nx);

    dim3 grid(B_SZ / 16), blk(256);
    lstm_layer_kernel<D_SZ, true><<<grid, blk, 0, stream>>>(
        xbf, Wih0, Whh0, bih0, bhh0, h0seq, nullptr);
    lstm_layer_kernel<H_SZ, false><<<grid, blk, 0, stream>>>(
        h0seq, Wih1, Whh1, bih1, bhh1, nullptr, hlast);
    fc_head_kernel<<<(B_SZ + 255) / 256, 256, 0, stream>>>(
        hlast, W1, b1, W2, b2, (float*)d_out);
}
